// PitchAndDurationExtractor_70566312673498
// MI455X (gfx1250) — matrix operand order repaired and checked by reading, not compiled, not run on hardware
//
#include <hip/hip_runtime.h>
#include <cstdint>
#include <cstddef>

// ---------------- constants ----------------
#define BB 32
#define TT 200        // T_TEXT
#define TM 1600       // T_MEL
#define HID 256
#define MEL 80
#define ATT 80
#define SPK 256
#define TEMPC 0.0005f
#define NEGV (-1e9f)

#define KEYS_FRAG_STRIDE (3 * 13 * 512)    // per-batch keysB fragment elements (19968)
#define Q_FRAG_TILE 1536                   // per 16-row m-tile: 3 ktiles * 512

typedef __attribute__((ext_vector_type(16))) __bf16 bf16x16;
typedef __attribute__((ext_vector_type(8)))  float  floatx8;

// ---------------- WMMA helpers (CDNA5 16x16x32 bf16) ----------------
__device__ __forceinline__ floatx8 wmma_bf16(bf16x16 a, bf16x16 b, floatx8 c) {
  return __builtin_amdgcn_wmma_f32_16x16x32_bf16(false, a, false, b, (short)0, c,
                                                 false, false);
}

__device__ __forceinline__ bf16x16 frag_ld(const __bf16* p) {
  return *(const bf16x16*)p;
}

// Fragment-slot offset for A-matrix element (m, k) in a fragment-ordered
// buffer [ktile][lane][16], contiguous K order:
// lanes 0-15 hold K {0..15}, lanes 16-31 hold K {16..31}.
__device__ __forceinline__ int a_frag_off(int m, int k) {
  int ko = k & 31;
  int hi = (ko >= 16) ? 1 : 0;
  int lane = m + (hi << 4);
  int e = ko & 15;
  return ((k >> 5) * 32 + lane) * 16 + e;
}

// Fragment-slot offset for B-matrix element (k, n) in buffer
// [ktile][ntile][lane][16]: lanes 0-15 -> K group kk+0..15 (col = lane),
// lanes 16-31 -> K group kk+16..31.
__device__ __forceinline__ int b_frag_off(int k, int n, int ntiles) {
  int kg = k & 31;
  int lane = (n & 15) + ((kg >= 16) ? 16 : 0);
  return (((k >> 5) * ntiles + (n >> 4)) * 32 + lane) * 16 + (kg & 15);
}

// ---------------- weight repack into B-fragment order ----------------
// src shape (N, C, 3); k = w*C + c; dst [KT][NTL][32][16]
__global__ void repack_k3_frag(const float* __restrict__ src, __bf16* __restrict__ dst,
                               int C, int KT, int NTL) {
  int idx = blockIdx.x * blockDim.x + threadIdx.x;
  if (idx >= KT * NTL * 512) return;
  int e = idx & 15, lane = (idx >> 4) & 31;
  int rest = idx >> 9;
  int nt = rest % NTL, kt = rest / NTL;
  int k = kt * 32 + ((lane >= 16) ? 16 : 0) + e;
  int n = nt * 16 + (lane & 15);
  float v = 0.f;
  if (k < 3 * C) {
    int w = k / C, c = k % C;
    v = src[(n * C + c) * 3 + w];
  }
  dst[idx] = (__bf16)v;
}

// src shape (N, Ksrc, 1)
__global__ void repack_1x1_frag(const float* __restrict__ src, __bf16* __restrict__ dst,
                                int Ksrc, int KT, int NTL) {
  int idx = blockIdx.x * blockDim.x + threadIdx.x;
  if (idx >= KT * NTL * 512) return;
  int e = idx & 15, lane = (idx >> 4) & 31;
  int rest = idx >> 9;
  int nt = rest % NTL, kt = rest / NTL;
  int k = kt * 32 + ((lane >= 16) ? 16 : 0) + e;
  int n = nt * 16 + (lane & 15);
  float v = (k < Ksrc) ? src[n * Ksrc + k] : 0.f;
  dst[idx] = (__bf16)v;
}

// ---------------- speaker projections ----------------
__global__ void spk_proj(const float* __restrict__ spk, const float* __restrict__ kw,
                         const float* __restrict__ kb, const float* __restrict__ qw,
                         const float* __restrict__ qb, float* __restrict__ keyAdd,
                         float* __restrict__ queryAdd) {
  int b = blockIdx.x, h = threadIdx.x;  // 256 threads
  float s = kb[h];
  for (int i = 0; i < SPK; ++i) s += spk[b * SPK + i] * kw[i * HID + h];
  keyAdd[b * HID + h] = s;
  if (h < MEL) {
    float s2 = qb[h];
    for (int i = 0; i < SPK; ++i) s2 += spk[b * SPK + i] * qw[i * MEL + h];
    queryAdd[b * MEL + h] = s2;
  }
}

// ---------------- zero fills (vectorized, b128 stores) ----------------
__global__ void zero_f32x4(float4* __restrict__ p, int n4) {
  int i = blockIdx.x * blockDim.x + threadIdx.x;
  if (i < n4) p[i] = make_float4(0.f, 0.f, 0.f, 0.f);
}
__global__ void zero_b16(uint4* __restrict__ p, int n16) {
  int i = blockIdx.x * blockDim.x + threadIdx.x;
  if (i < n16) {
    uint4 z = {0u, 0u, 0u, 0u};
    p[i] = z;
  }
}

// ---------------- keys: conv3(256->512,relu) + 1x1(512->80) ----------------
__global__ __launch_bounds__(128) void keys_kernel(
    const float* __restrict__ hs, const float* __restrict__ keyAdd,
    const __bf16* __restrict__ W1b, const float* __restrict__ kb1,
    const __bf16* __restrict__ W2b, const float* __restrict__ kb2,
    __bf16* __restrict__ keysBfrag) {
  __shared__ __bf16 Afrag[24 * 512];  // K=768, A-fragment order
  __shared__ __bf16 Hfrag[16 * 512];  // K=512, A-fragment order
  int tid = threadIdx.x, lane = tid & 31, wv = tid >> 5;
  int tile = blockIdx.x;  // 400 tiles of 16 positions over B*TT

  for (int idx = tid; idx < 16 * 768; idx += 128) {
    int r = idx / 768, k = idx % 768;
    int w = k / HID, c = k % HID;
    int p = tile * 16 + r, b = p / TT, t = p % TT;
    int tw = t + w - 1;
    float v = 0.f;
    if (tw >= 0 && tw < TT) v = hs[(b * TT + tw) * HID + c] + keyAdd[b * HID + c];
    Afrag[a_frag_off(r, k)] = (__bf16)v;
  }
  __syncthreads();

  // stage 1: [16x768] x [768x512] -> relu -> Hfrag
  for (int nt = wv; nt < 32; nt += 4) {
    floatx8 acc = {};
    for (int kt = 0; kt < 24; ++kt) {
      bf16x16 af = frag_ld(&Afrag[(kt * 32 + lane) * 16]);
      bf16x16 bf = frag_ld(&W1b[((kt * 32 + nt) * 32 + lane) * 16]);
      acc = wmma_bf16(af, bf, acc);
    }
    int n = nt * 16 + (lane & 15);
    float bias = kb1[n];
    int mh = (lane < 16) ? 0 : 8;
#pragma unroll
    for (int r = 0; r < 8; ++r)
      Hfrag[a_frag_off(r + mh, n)] = (__bf16)fmaxf(acc[r] + bias, 0.f);
  }
  __syncthreads();

  // stage 2: [16x512] x [512x80] -> keysBfrag (fragment order, n-dim = t)
  for (int nt = wv; nt < 5; nt += 4) {
    floatx8 acc = {};
    for (int kt = 0; kt < 16; ++kt) {
      bf16x16 af = frag_ld(&Hfrag[(kt * 32 + lane) * 16]);
      bf16x16 bf = frag_ld(&W2b[((kt * 5 + nt) * 32 + lane) * 16]);
      acc = wmma_bf16(af, bf, acc);
    }
    int n = nt * 16 + (lane & 15);  // output channel c
    float bias = kb2[n];
    int mh = (lane < 16) ? 0 : 8;
#pragma unroll
    for (int r = 0; r < 8; ++r) {
      int p = tile * 16 + r + mh, b = p / TT, t = p % TT;
      keysBfrag[(size_t)b * KEYS_FRAG_STRIDE + b_frag_off(n, t, 13)] =
          (__bf16)(acc[r] + bias);
    }
  }
}

// ---------------- queries: conv3(80->160,relu)+1x1(160->80,relu)+1x1(80->80) ----
__global__ __launch_bounds__(128) void queries_kernel(
    const float* __restrict__ mels, const float* __restrict__ queryAdd,
    const __bf16* __restrict__ QW1b, const float* __restrict__ qb1,
    const __bf16* __restrict__ QW2b, const float* __restrict__ qb2,
    const __bf16* __restrict__ QW3b, const float* __restrict__ qb3,
    __bf16* __restrict__ Qfrag) {
  __shared__ __bf16 Afrag[8 * 512];   // K=256 (padded from 240)
  __shared__ __bf16 H1f[5 * 512];     // K=160
  __shared__ __bf16 H2f[3 * 512];     // K=96 (padded from 80)
  int tid = threadIdx.x, lane = tid & 31, wv = tid >> 5;
  int tile = blockIdx.x;  // 3200 tiles over B*TM (tiles never cross batch)

  for (int idx = tid; idx < 3 * 512; idx += 128) H2f[idx] = (__bf16)0.f;
  for (int idx = tid; idx < 16 * 256; idx += 128) {
    int r = idx / 256, k = idx % 256;
    float v = 0.f;
    if (k < 240) {
      int w = k / MEL, c = k % MEL;
      int p = tile * 16 + r, b = p / TM, m = p % TM;
      int tw = m + w - 1;
      if (tw >= 0 && tw < TM)
        v = mels[(b * TM + tw) * MEL + c] + queryAdd[b * MEL + c];
    }
    Afrag[a_frag_off(r, k)] = (__bf16)v;
  }
  __syncthreads();

  for (int nt = wv; nt < 10; nt += 4) {  // N=160, K=256
    floatx8 acc = {};
    for (int kt = 0; kt < 8; ++kt) {
      bf16x16 af = frag_ld(&Afrag[(kt * 32 + lane) * 16]);
      bf16x16 bf = frag_ld(&QW1b[((kt * 10 + nt) * 32 + lane) * 16]);
      acc = wmma_bf16(af, bf, acc);
    }
    int n = nt * 16 + (lane & 15);
    float bias = qb1[n];
    int mh = (lane < 16) ? 0 : 8;
#pragma unroll
    for (int r = 0; r < 8; ++r)
      H1f[a_frag_off(r + mh, n)] = (__bf16)fmaxf(acc[r] + bias, 0.f);
  }
  __syncthreads();

  for (int nt = wv; nt < 5; nt += 4) {  // N=80, K=160
    floatx8 acc = {};
    for (int kt = 0; kt < 5; ++kt) {
      bf16x16 af = frag_ld(&H1f[(kt * 32 + lane) * 16]);
      bf16x16 bf = frag_ld(&QW2b[((kt * 5 + nt) * 32 + lane) * 16]);
      acc = wmma_bf16(af, bf, acc);
    }
    int n = nt * 16 + (lane & 15);
    float bias = qb2[n];
    int mh = (lane < 16) ? 0 : 8;
#pragma unroll
    for (int r = 0; r < 8; ++r)
      H2f[a_frag_off(r + mh, n)] = (__bf16)fmaxf(acc[r] + bias, 0.f);
  }
  __syncthreads();

  for (int nt = wv; nt < 5; nt += 4) {  // N=80, K=96
    floatx8 acc = {};
    for (int kt = 0; kt < 3; ++kt) {
      bf16x16 af = frag_ld(&H2f[(kt * 32 + lane) * 16]);
      bf16x16 bf = frag_ld(&QW3b[((kt * 5 + nt) * 32 + lane) * 16]);
      acc = wmma_bf16(af, bf, acc);
    }
    int n = nt * 16 + (lane & 15);
    float bias = qb3[n];
    int mh = (lane < 16) ? 0 : 8;
#pragma unroll
    for (int r = 0; r < 8; ++r) {
      // tile == b*100 + mtile; write A-fragment-ordered Q for the score GEMM
      Qfrag[(size_t)tile * Q_FRAG_TILE + a_frag_off(r + mh, n)] =
          (__bf16)(acc[r] + bias);
    }
  }
}

// ---------------- q2 / k2 ----------------
__global__ void compute_q2(const __bf16* __restrict__ Qfrag, float* __restrict__ q2) {
  int idx = blockIdx.x * blockDim.x + threadIdx.x;
  if (idx >= BB * TM) return;
  int b = idx / TM, m = idx % TM;
  const __bf16* Qf = Qfrag + (size_t)(b * 100 + (m >> 4)) * Q_FRAG_TILE;
  int r = m & 15;
  float s = 0.f;
  for (int c = 0; c < ATT; ++c) {
    float v = (float)Qf[a_frag_off(r, c)];
    s += v * v;
  }
  q2[idx] = s;
}

__global__ void compute_k2(const __bf16* __restrict__ keysBfrag,
                           float* __restrict__ k2) {
  int idx = blockIdx.x * blockDim.x + threadIdx.x;
  if (idx >= BB * TT) return;
  int b = idx / TT, t = idx % TT;
  const __bf16* Kf = keysBfrag + (size_t)b * KEYS_FRAG_STRIDE;
  float s = 0.f;
  for (int c = 0; c < ATT; ++c) {
    float v = (float)Kf[b_frag_off(c, t, 13)];
    s += v * v;
  }
  k2[idx] = s;
}

// ---------------- score + softmaxes ----------------
__global__ __launch_bounds__(128) void score_kernel(
    const __bf16* __restrict__ Qfrag, const __bf16* __restrict__ keysBfrag,
    const float* __restrict__ q2, const float* __restrict__ k2,
    const float* __restrict__ priors, const int* __restrict__ plens,
    float* __restrict__ out_logprob, float* __restrict__ out_soft) {
  __shared__ float Sbuf[16 * 208];
  int tid = threadIdx.x, lane = tid & 31, wv = tid >> 5;
  int blk = blockIdx.x, b = blk / 100, mtile = blk % 100;
  const __bf16* Qf = Qfrag + (size_t)blk * Q_FRAG_TILE;
  const __bf16* Kf = keysBfrag + (size_t)b * KEYS_FRAG_STRIDE;

  for (int nt = wv; nt < 13; nt += 4) {  // 13 t-tiles cover 208 >= 200
    floatx8 acc = {};
    for (int kt = 0; kt < 3; ++kt) {
      bf16x16 af = frag_ld(&Qf[(kt * 32 + lane) * 16]);
      bf16x16 bf = frag_ld(&Kf[((kt * 13 + nt) * 32 + lane) * 16]);
      acc = wmma_bf16(af, bf, acc);
    }
    int n = nt * 16 + (lane & 15);
    int mh = (lane < 16) ? 0 : 8;
#pragma unroll
    for (int r = 0; r < 8; ++r) Sbuf[(r + mh) * 208 + n] = acc[r];
  }
  __syncthreads();

  // softmax phase: 8 threads per row, butterfly reductions within 8-lane groups
  {
    int r = tid >> 3;  // 0..15
    int g = tid & 7;   // 0..7
    int m = mtile * 16 + r;
    float* row = &Sbuf[r * 208];
    float q2v = q2[b * TM + m];
    int plen = plens[b];

    float mx = -INFINITY;
    for (int t = g; t < TT; t += 8) {
      float s = -TEMPC * (q2v + k2[b * TT + t] - 2.f * row[t]);
      row[t] = s;
      mx = fmaxf(mx, s);
    }
    mx = fmaxf(mx, __shfl_xor(mx, 1));
    mx = fmaxf(mx, __shfl_xor(mx, 2));
    mx = fmaxf(mx, __shfl_xor(mx, 4));

    float sum = 0.f;
    for (int t = g; t < TT; t += 8) sum += expf(row[t] - mx);
    sum += __shfl_xor(sum, 1);
    sum += __shfl_xor(sum, 2);
    sum += __shfl_xor(sum, 4);
    float lse = mx + logf(sum);

    float mx2 = -INFINITY;
    for (int t = g; t < TT; t += 8) {
      float lp = row[t] - lse + logf(priors[((size_t)(b * TT + t)) * TM + m] + 1e-8f);
      out_logprob[((size_t)(b * TM + m)) * TT + t] = lp;
      row[t] = lp;
      if (t < plen) mx2 = fmaxf(mx2, lp);
    }
    mx2 = fmaxf(mx2, __shfl_xor(mx2, 1));
    mx2 = fmaxf(mx2, __shfl_xor(mx2, 2));
    mx2 = fmaxf(mx2, __shfl_xor(mx2, 4));

    float sum2 = 0.f;
    for (int t = g; t < plen; t += 8) sum2 += expf(row[t] - mx2);
    sum2 += __shfl_xor(sum2, 1);
    sum2 += __shfl_xor(sum2, 2);
    sum2 += __shfl_xor(sum2, 4);
    float inv = 1.f / sum2;

    for (int t = g; t < TT; t += 8) {
      float v = (t < plen) ? expf(row[t] - mx2) * inv : 0.f;
      out_soft[((size_t)(b * TM + m)) * TT + t] = v;
    }
  }
}

// -------- MAS forward (one wave32 per batch, register DP + prefetch) --------
__global__ __launch_bounds__(32) void mas_forward(const float* __restrict__ soft,
                                                  const int* __restrict__ plens,
                                                  uint32_t* __restrict__ choices) {
  int b = blockIdx.x, lane = threadIdx.x;
  int tlen = plens[b];
  const float* base = soft + (size_t)b * TM * TT;
  float prev[7];
#pragma unroll
  for (int c = 0; c < 7; ++c) {
    int j = c * 32 + lane;
    prev[c] = (j == 0) ? logf(fmaxf(base[0], 1e-30f)) : NEGV;
  }
  if (lane < 7) choices[(size_t)(b * TM) * 7 + lane] = 0u;

  float nxt[7];
#pragma unroll
  for (int c = 0; c < 7; ++c) {
    int j = c * 32 + lane;
    nxt[c] = (j < TT) ? base[TT + j] : 0.f;
  }

  for (int i = 1; i < TM; ++i) {
    float cur[7];
#pragma unroll
    for (int c = 0; c < 7; ++c) cur[c] = nxt[c];
    if (i + 1 < TM) {
#pragma unroll
      for (int c = 0; c < 7; ++c) {
        int j = c * 32 + lane;
        nxt[c] = (j < TT) ? base[(size_t)(i + 1) * TT + j] : 0.f;
      }
    }
    float last[7];
#pragma unroll
    for (int c = 0; c < 7; ++c) last[c] = __shfl(prev[c], 31);
    uint32_t w0 = 0, w1 = 0, w2 = 0, w3 = 0, w4 = 0, w5 = 0, w6 = 0;
#pragma unroll
    for (int c = 0; c < 7; ++c) {
      int j = c * 32 + lane;
      float row = (j < tlen) ? logf(fmaxf(cur[c], 1e-30f)) : NEGV;
      float sh = __shfl_up(prev[c], 1);
      if (lane == 0) sh = (c == 0) ? NEGV : last[c - 1];
      bool ch = (sh >= prev[c]);
      prev[c] = row + fmaxf(prev[c], sh);
      uint32_t bal = (uint32_t)__ballot(ch);
      if (c == 0) w0 = bal; else if (c == 1) w1 = bal; else if (c == 2) w2 = bal;
      else if (c == 3) w3 = bal; else if (c == 4) w4 = bal; else if (c == 5) w5 = bal;
      else w6 = bal;
    }
    if (lane == 0) {
      uint32_t* dst = &choices[(size_t)(b * TM + i) * 7];
      dst[0] = w0; dst[1] = w1; dst[2] = w2; dst[3] = w3;
      dst[4] = w4; dst[5] = w5; dst[6] = w6;
    }
  }
}

// ---------------- backtrack + durations + avg pitch ----------------
__global__ __launch_bounds__(64) void backtrack_kernel(
    const uint32_t* __restrict__ choices, const int* __restrict__ plens,
    const int* __restrict__ mlens, const float* __restrict__ pitches,
    float* __restrict__ out_hard, float* __restrict__ out_dur,
    float* __restrict__ out_avg) {
  __shared__ float dur[TT];
  __shared__ float psum[TT];
  int b = blockIdx.x, tid = threadIdx.x;
  for (int t = tid; t < TT; t += 64) { dur[t] = 0.f; psum[t] = 0.f; }
  __syncthreads();
  if (tid == 0) {
    int tlen = plens[b], mlen = mlens[b];
    int j = tlen - 1;
    for (int i = TM - 1; i >= 0; --i) {
      if (i == mlen - 1) j = tlen - 1;
      if (i < mlen) {
        out_hard[((size_t)(b * TM + i)) * TT + j] = 1.0f;
        dur[j] += 1.f;
        psum[j] += pitches[b * TM + i];
      }
      uint32_t w = choices[(size_t)(b * TM + i) * 7 + (j >> 5)];
      j -= (int)((w >> (j & 31)) & 1u);
      if (j < 0) j = 0;
    }
  }
  __syncthreads();
  for (int t = tid; t < TT; t += 64) {
    float d = dur[t];
    out_dur[b * TT + t] = d;
    out_avg[b * TT + t] = (d > 0.f) ? psum[t] / fmaxf(d, 1.f) : 0.f;
  }
}

// ---------------- launch ----------------
extern "C" void kernel_launch(void* const* d_in, const int* in_sizes, int n_in,
                              void* d_out, int out_size, void* d_ws, size_t ws_size,
                              hipStream_t stream) {
  const float* hs      = (const float*)d_in[0];
  const float* pitches = (const float*)d_in[1];
  const float* mels    = (const float*)d_in[2];
  const float* priors  = (const float*)d_in[3];
  const float* spk     = (const float*)d_in[4];
  const int*   plens   = (const int*)d_in[5];
  const int*   mlens   = (const int*)d_in[6];
  const float* key_w   = (const float*)d_in[7];
  const float* key_b   = (const float*)d_in[8];
  const float* qry_w   = (const float*)d_in[9];
  const float* qry_b   = (const float*)d_in[10];
  const float* kw1 = (const float*)d_in[11];
  const float* kb1 = (const float*)d_in[12];
  const float* kw2 = (const float*)d_in[13];
  const float* kb2 = (const float*)d_in[14];
  const float* qw1 = (const float*)d_in[15];
  const float* qb1 = (const float*)d_in[16];
  const float* qw2 = (const float*)d_in[17];
  const float* qb2 = (const float*)d_in[18];
  const float* qw3 = (const float*)d_in[19];
  const float* qb3 = (const float*)d_in[20];

  char* ws = (char*)d_ws;
  size_t off = 0;
  auto alloc = [&](size_t bytes) {
    void* p = ws + off;
    off = (off + bytes + 255) & ~(size_t)255;
    return p;
  };
  float*    keyAdd    = (float*)alloc(BB * HID * 4);
  float*    queryAdd  = (float*)alloc(BB * MEL * 4);
  __bf16*   W1b       = (__bf16*)alloc((size_t)24 * 32 * 512 * 2);  // 768x512
  __bf16*   W2b       = (__bf16*)alloc((size_t)16 * 5 * 512 * 2);   // 512x80
  __bf16*   QW1b      = (__bf16*)alloc((size_t)8 * 10 * 512 * 2);   // 256x160
  __bf16*   QW2b      = (__bf16*)alloc((size_t)5 * 5 * 512 * 2);    // 160x80
  __bf16*   QW3b      = (__bf16*)alloc((size_t)3 * 5 * 512 * 2);    // 96x80
  __bf16*   keysBfrag = (__bf16*)alloc((size_t)BB * KEYS_FRAG_STRIDE * 2);
  __bf16*   Qfrag     = (__bf16*)alloc((size_t)BB * 100 * Q_FRAG_TILE * 2);
  float*    k2        = (float*)alloc(BB * TT * 4);
  float*    q2        = (float*)alloc((size_t)BB * TM * 4);
  uint32_t* choices   = (uint32_t*)alloc((size_t)BB * TM * 7 * 4);

  float* out         = (float*)d_out;
  float* out_avg     = out;             // (B,200,1)
  float* out_dur     = out + 6400;      // (B,200)
  float* out_soft    = out + 12800;     // (B,1,1600,200)
  float* out_hard    = out + 10252800;
  float* out_logprob = out + 20492800;

  // 1) weight repacks into fragment order
  repack_k3_frag<<<(24 * 32 * 512 + 255) / 256, 256, 0, stream>>>(kw1, W1b, 256, 24, 32);
  repack_k3_frag<<<(8 * 10 * 512 + 255) / 256, 256, 0, stream>>>(qw1, QW1b, 80, 8, 10);
  repack_1x1_frag<<<(16 * 5 * 512 + 255) / 256, 256, 0, stream>>>(kw2, W2b, 512, 16, 5);
  repack_1x1_frag<<<(5 * 5 * 512 + 255) / 256, 256, 0, stream>>>(qw2, QW2b, 160, 5, 5);
  repack_1x1_frag<<<(3 * 5 * 512 + 255) / 256, 256, 0, stream>>>(qw3, QW3b, 80, 3, 5);

  // 2) speaker projections + zero pads of fragment buffers (16B stores)
  spk_proj<<<BB, 256, 0, stream>>>(spk, key_w, key_b, qry_w, qry_b, keyAdd, queryAdd);
  {
    int nk16 = (BB * KEYS_FRAG_STRIDE * 2) / 16;
    zero_b16<<<(nk16 + 255) / 256, 256, 0, stream>>>((uint4*)keysBfrag, nk16);
    int nq16 = (BB * 100 * Q_FRAG_TILE * 2) / 16;
    zero_b16<<<(nq16 + 255) / 256, 256, 0, stream>>>((uint4*)Qfrag, nq16);
  }

  // 3) keys conv stack (WMMA)
  keys_kernel<<<(BB * TT) / 16, 128, 0, stream>>>(hs, keyAdd, W1b, kb1, W2b, kb2,
                                                  keysBfrag);

  // 4) queries conv stack (WMMA)
  queries_kernel<<<(BB * TM) / 16, 128, 0, stream>>>(mels, queryAdd, QW1b, qb1,
                                                     QW2b, qb2, QW3b, qb3, Qfrag);

  // 5) squared norms
  compute_q2<<<(BB * TM + 255) / 256, 256, 0, stream>>>(Qfrag, q2);
  compute_k2<<<(BB * TT + 255) / 256, 256, 0, stream>>>(keysBfrag, k2);

  // 6) scores + softmaxes (WMMA for Q.K^T)
  score_kernel<<<BB * 100, 128, 0, stream>>>(Qfrag, keysBfrag, q2, k2, priors, plens,
                                             out_logprob, out_soft);

  // 7) zero attn_hard (b128 stores)
  {
    int n4 = (BB * TM * TT) / 4;
    zero_f32x4<<<(n4 + 255) / 256, 256, 0, stream>>>((float4*)out_hard, n4);
  }

  // 8) MAS forward DP
  mas_forward<<<BB, 32, 0, stream>>>(out_soft, plens, choices);

  // 9) backtrack + durations + avg pitch
  backtrack_kernel<<<BB, 64, 0, stream>>>(choices, plens, mlens, pitches,
                                          out_hard, out_dur, out_avg);
}
